// CausalSelfAttention_30477087932974
// MI455X (gfx1250) — matrix operand order repaired and checked by reading, not compiled, not run on hardware
//
#include <hip/hip_runtime.h>
#include <hip/hip_bf16.h>

// ---------------------------------------------------------------------------
// CDNA5 (gfx1250) prefix-LM attention block, bf16 WMMA everywhere.
// B=2, T=2048, C=1024, H=16, D=64, prefix_len=128 (from reference setup).
// Round 2: double-buffered GEMM staging, scalar q/k/v select, async
// global->LDS (ASYNCcnt) staging in the projection GEMM.
// ---------------------------------------------------------------------------

typedef __bf16 bf16_t;
typedef __attribute__((ext_vector_type(8)))  __bf16 v8bf;
typedef __attribute__((ext_vector_type(16))) __bf16 v16bf;
typedef __attribute__((ext_vector_type(8)))  float  v8f;

#define B_   2
#define T_   2048
#define C_   1024
#define H_   16
#define D_   64
#define PFX_ 128

__device__ __forceinline__ v16bf cat16(v8bf lo, v8bf hi) {
  return __builtin_shufflevector(lo, hi, 0,1,2,3,4,5,6,7,8,9,10,11,12,13,14,15);
}

__device__ __forceinline__ v8f wmma_bf16(v16bf a, v16bf b, v8f c) {
  // D = A(16x32 bf16) * B(32x16 bf16) + C(16x16 f32)
  return __builtin_amdgcn_wmma_f32_16x16x32_bf16(false, a, false, b,
                                                 (short)0, c, false, false);
}

// Async global->LDS copy, 16 bytes per lane (GLOBAL_LOAD_ASYNC_TO_LDS_B128).
// lds_byte_off is a per-lane LDS byte address (dynamic LDS starts at offset 0).
__device__ __forceinline__ void async_g2l_b128(unsigned lds_byte_off,
                                               const void* gaddr) {
  asm volatile("global_load_async_to_lds_b128 %0, %1, off"
               :: "v"(lds_byte_off), "v"(gaddr) : "memory");
}
__device__ __forceinline__ void wait_async0() {
  asm volatile("s_wait_asynccnt 0" ::: "memory");
}

// ---------------------------------------------------------------------------
// Kernel 1: qkv = x @ Wqkv + bqkv   (M=4096, N=3072, K=1024)
// Double-buffered LDS; output scattered bf16: q,k -> [B,H,T,D]; v -> [B,H,D,T]
// ---------------------------------------------------------------------------
__global__ __launch_bounds__(256) void qkv_gemm_kernel(
    const float* __restrict__ x, const float* __restrict__ W,
    const float* __restrict__ bias,
    bf16_t* __restrict__ qb, bf16_t* __restrict__ kb, bf16_t* __restrict__ vtb)
{
  __shared__ __align__(16) bf16_t As[2][128 * 32];   // [m][k]
  __shared__ __align__(16) bf16_t Bs[2][128 * 32];   // [n][k] (transposed)

  const int tid  = threadIdx.x;
  const int wave = tid >> 5, lane = tid & 31;
  const int l15  = lane & 15, half = lane >> 4;
  const int m0 = (blockIdx.x & 31) * 128;          // 32 M-tiles
  const int n0 = (blockIdx.x >> 5) * 128;          // 24 N-tiles
  const int wm = wave & 3, wn = wave >> 2;         // wave tile 32x64

  const int a_row = tid >> 1, a_k = (tid & 1) * 16;
  const int b_n   = tid & 127, b_k = (tid >> 7) * 16;

  v8f acc[2][4] = {};

  // ---- prologue: stage tile k0=0 into buffer 0 ----
  {
    const float* ap = x + (size_t)(m0 + a_row) * C_ + a_k;
    v8f alo = *(const v8f*)ap;
    v8f ahi = *(const v8f*)(ap + 8);
    *(v8bf*)&As[0][a_row * 32 + a_k]     = __builtin_convertvector(alo, v8bf);
    *(v8bf*)&As[0][a_row * 32 + a_k + 8] = __builtin_convertvector(ahi, v8bf);

    const float* bp = W + (size_t)b_k * 3072 + n0 + b_n;
    float bv[16];
#pragma unroll
    for (int j = 0; j < 16; ++j) bv[j] = bp[(size_t)j * 3072];
    v8bf blo, bhi;
#pragma unroll
    for (int j = 0; j < 8; ++j) { blo[j] = (bf16_t)bv[j]; bhi[j] = (bf16_t)bv[8 + j]; }
    *(v8bf*)&Bs[0][b_n * 32 + b_k]     = blo;
    *(v8bf*)&Bs[0][b_n * 32 + b_k + 8] = bhi;
  }
  __syncthreads();

  for (int k0 = 0; k0 < C_; k0 += 32) {
    const int  cur = (k0 >> 5) & 1;
    const bool nxt = (k0 + 32) < C_;

    // ---- issue next tile's global loads (overlap with WMMA below) ----
    v8f alo, ahi; float bv[16];
    if (nxt) {
      const float* ap = x + (size_t)(m0 + a_row) * C_ + (k0 + 32) + a_k;
      alo = *(const v8f*)ap;
      ahi = *(const v8f*)(ap + 8);
      const float* bp = W + (size_t)(k0 + 32 + b_k) * 3072 + n0 + b_n;
#pragma unroll
      for (int j = 0; j < 16; ++j) bv[j] = bp[(size_t)j * 3072];
    }

    // ---- compute on current buffer ----
    v16bf afr[2];
#pragma unroll
    for (int mi = 0; mi < 2; ++mi) {
      const bf16_t* ar = &As[cur][(wm * 32 + mi * 16 + l15) * 32];
      afr[mi] = cat16(*(const v8bf*)(ar + half * 8),
                      *(const v8bf*)(ar + 16 + half * 8));
    }
#pragma unroll
    for (int ni = 0; ni < 4; ++ni) {
      const bf16_t* br = &Bs[cur][(wn * 64 + ni * 16 + l15) * 32 + half * 8];
      v16bf bfr = cat16(*(const v8bf*)br, *(const v8bf*)(br + 16));
#pragma unroll
      for (int mi = 0; mi < 2; ++mi)
        acc[mi][ni] = wmma_bf16(afr[mi], bfr, acc[mi][ni]);
    }

    // ---- stash next tile into the other buffer ----
    if (nxt) {
      *(v8bf*)&As[cur ^ 1][a_row * 32 + a_k]     = __builtin_convertvector(alo, v8bf);
      *(v8bf*)&As[cur ^ 1][a_row * 32 + a_k + 8] = __builtin_convertvector(ahi, v8bf);
      v8bf blo, bhi;
#pragma unroll
      for (int j = 0; j < 8; ++j) { blo[j] = (bf16_t)bv[j]; bhi[j] = (bf16_t)bv[8 + j]; }
      *(v8bf*)&Bs[cur ^ 1][b_n * 32 + b_k]     = blo;
      *(v8bf*)&Bs[cur ^ 1][b_n * 32 + b_k + 8] = bhi;
    }
    __syncthreads();
  }

  // ---- epilogue: bias add + scatter (which is block-uniform scalar) ----
  const int which = n0 >> 10;          // 0=q 1=k 2=v
#pragma unroll
  for (int ni = 0; ni < 4; ++ni) {
    const int n = n0 + wn * 64 + ni * 16 + l15;
    const int c = n & 1023, h = c >> 6, d = c & 63;
    const float bvv = bias[n];
#pragma unroll
    for (int mi = 0; mi < 2; ++mi) {
#pragma unroll
      for (int g = 0; g < 8; ++g) {
        const int m = m0 + wm * 32 + mi * 16 + g + 8 * half;
        const int bb = m >> 11, t = m & (T_ - 1);
        const bf16_t val = (bf16_t)(acc[mi][ni][g] + bvv);
        if (which == 0)
          qb[(((size_t)bb * H_ + h) * T_ + t) * D_ + d] = val;
        else if (which == 1)
          kb[(((size_t)bb * H_ + h) * T_ + t) * D_ + d] = val;
        else
          vtb[(((size_t)bb * H_ + h) * D_ + d) * T_ + t] = val;
      }
    }
  }
}

// ---------------------------------------------------------------------------
// Kernel 2: per (b,h,16-row q block) attention with prefix-LM mask.
// LDS: S[16][2048] fp32 (128KB) + P[16][2048] bf16 (64KB) + Y[16][64] + sums.
// ---------------------------------------------------------------------------
__global__ __launch_bounds__(256) void attn_kernel(
    const bf16_t* __restrict__ qb, const bf16_t* __restrict__ kb,
    const bf16_t* __restrict__ vtb, bf16_t* __restrict__ yb)
{
  extern __shared__ __align__(16) char smem[];
  float*  S    = (float*)smem;                                // [16][2048]
  bf16_t* Pl   = (bf16_t*)(smem + 16 * T_ * 4);               // [16][2048]
  float*  Y    = (float*)(smem + 16 * T_ * 4 + 16 * T_ * 2);  // [16][64]
  float*  Ssum = Y + 16 * 64;                                 // [16]

  const int tid  = threadIdx.x;
  const int wave = tid >> 5, lane = tid & 31;
  const int l15  = lane & 15, half = lane >> 4;
  const int qblk = blockIdx.x & 127;
  const int bh   = blockIdx.x >> 7;                // b*H + h
  const int i0   = qblk * 16;
  const size_t hdoff = (size_t)bh * T_ * D_;

  // prefix-LM: rows < PFX see all T keys; rows >= PFX see keys [0, i].
  const int ktend = (i0 < PFX_) ? (T_ / 64) : ((i0 + 16 + 63) >> 6);

  for (int e = tid; e < 16 * 64; e += 256) Y[e] = 0.f;

  // Q fragments (invariant over key tiles): 2 k-steps across D=64
  v16bf qf[2];
  {
    const bf16_t* qr = qb + hdoff + (size_t)(i0 + l15) * D_;
#pragma unroll
    for (int s = 0; s < 2; ++s)
      qf[s] = cat16(*(const v8bf*)(qr + s * 32 + half * 8),
                    *(const v8bf*)(qr + s * 32 + 16 + half * 8));
  }

  // ---- S = Q K^T over key tiles (waves split tiles) ----
  for (int kt = wave; kt < ktend; kt += 8) {
    const int j0 = kt * 64;
    v8f sa[4] = {};
#pragma unroll
    for (int s = 0; s < 2; ++s) {
#pragma unroll
      for (int nf = 0; nf < 4; ++nf) {
        const bf16_t* kr = kb + hdoff + (size_t)(j0 + nf * 16 + l15) * D_
                         + s * 32 + half * 8;
        sa[nf] = wmma_bf16(qf[s], cat16(*(const v8bf*)kr, *(const v8bf*)(kr + 16)),
                           sa[nf]);
      }
    }
#pragma unroll
    for (int nf = 0; nf < 4; ++nf) {
      const int n = j0 + nf * 16 + l15;
#pragma unroll
      for (int g = 0; g < 8; ++g) S[(g + 8 * half) * T_ + n] = sa[nf][g];
    }
  }
  __syncthreads();

  // ---- masked softmax (unnormalized; per-row sum kept) ----
  const float c2 = 0.125f * 1.44269504088896f;   // scale * log2(e)
  for (int r = wave; r < 16; r += 8) {
    const int i = i0 + r;
    const int jmax = (i < PFX_) ? T_ : (i + 1);
    const int jpad = ktend * 64;
    float* srow = S + r * T_;
    bf16_t* prow = Pl + r * T_;
    float mx = -3.0e38f;
    for (int j = lane; j < jmax; j += 32) mx = fmaxf(mx, srow[j]);
#pragma unroll
    for (int o = 16; o > 0; o >>= 1) mx = fmaxf(mx, __shfl_xor(mx, o, 32));
    float sum = 0.f;
    for (int j = lane; j < jpad; j += 32) {
      float e = 0.f;
      if (j < jmax) e = exp2f((srow[j] - mx) * c2);
      prow[j] = (bf16_t)e;
      sum += e;
    }
#pragma unroll
    for (int o = 16; o > 0; o >>= 1) sum += __shfl_xor(sum, o, 32);
    if (lane == 0) Ssum[r] = sum;
  }
  __syncthreads();

  // ---- Y = P @ V (waves split key tiles; V stored transposed) ----
  v8f yacc[4] = {};
  for (int kt = wave; kt < ktend; kt += 8) {
    const int j0 = kt * 64;
#pragma unroll
    for (int s = 0; s < 2; ++s) {
      const bf16_t* pr = Pl + l15 * T_ + j0 + s * 32;
      v16bf pf = cat16(*(const v8bf*)(pr + half * 8),
                       *(const v8bf*)(pr + 16 + half * 8));
#pragma unroll
      for (int nf = 0; nf < 4; ++nf) {
        const bf16_t* vr = vtb + (size_t)bh * D_ * T_
                         + (size_t)(nf * 16 + l15) * T_ + j0 + s * 32 + half * 8;
        yacc[nf] = wmma_bf16(pf, cat16(*(const v8bf*)vr, *(const v8bf*)(vr + 16)),
                             yacc[nf]);
      }
    }
  }
#pragma unroll
  for (int nf = 0; nf < 4; ++nf) {
    const int d = nf * 16 + l15;
#pragma unroll
    for (int g = 0; g < 8; ++g)
      atomicAdd(&Y[(g + 8 * half) * 64 + d], yacc[nf][g]);
  }
  __syncthreads();

  // ---- normalize + write bf16 y rows into [B*T, C] ----
  const int bb = bh >> 4, h = bh & 15;
  for (int e = tid; e < 16 * 64; e += 256) {
    const int r = e >> 6, d = e & 63;
    const float val = Y[e] / Ssum[r];
    yb[((size_t)bb * T_ + i0 + r) * C_ + h * 64 + d] = (bf16_t)val;
  }
}

// ---------------------------------------------------------------------------
// Kernel 3: out = y @ Wproj + bproj   (M=4096, N=1024, K=1024), fp32 out.
// A operand is already bf16 -> staged with GLOBAL_LOAD_ASYNC_TO_LDS_B128.
// Dynamic LDS layout (starts at offset 0):
//   A[2][128*32] bf16 @ 0      (2 x 8KB)
//   B[2][128*32] bf16 @ 16384  (2 x 8KB)
// ---------------------------------------------------------------------------
__global__ __launch_bounds__(256) void proj_gemm_kernel(
    const bf16_t* __restrict__ yb, const float* __restrict__ W,
    const float* __restrict__ bias, float* __restrict__ out)
{
  extern __shared__ __align__(16) char smem3[];
  bf16_t* Abuf = (bf16_t*)smem3;                  // [2][4096]
  bf16_t* Bbuf = (bf16_t*)(smem3 + 16384);        // [2][4096]

  const int tid  = threadIdx.x;
  const int wave = tid >> 5, lane = tid & 31;
  const int l15  = lane & 15, half = lane >> 4;
  const int m0 = (blockIdx.x & 31) * 128;          // 32 M-tiles
  const int n0 = (blockIdx.x >> 5) * 128;          // 8 N-tiles
  const int wm = wave & 3, wn = wave >> 2;

  const int a_row = tid >> 1, a_k = (tid & 1) * 16;
  const int b_n   = tid & 127, b_k = (tid >> 7) * 16;

  // per-lane LDS byte offset of this thread's 32B A-slot within a buffer
  const unsigned a_lds = (unsigned)((a_row * 32 + a_k) * 2);

  v8f acc[2][4] = {};

  // ---- prologue: stage tile 0 ----
  {
    const bf16_t* ap = yb + (size_t)(m0 + a_row) * C_ + a_k;
    async_g2l_b128(a_lds,      ap);        // 16B
    async_g2l_b128(a_lds + 16, ap + 8);    // 16B

    const float* bp = W + (size_t)b_k * C_ + n0 + b_n;
    float bv[16];
#pragma unroll
    for (int j = 0; j < 16; ++j) bv[j] = bp[(size_t)j * C_];
    v8bf blo, bhi;
#pragma unroll
    for (int j = 0; j < 8; ++j) { blo[j] = (bf16_t)bv[j]; bhi[j] = (bf16_t)bv[8 + j]; }
    *(v8bf*)&Bbuf[b_n * 32 + b_k]     = blo;
    *(v8bf*)&Bbuf[b_n * 32 + b_k + 8] = bhi;
    wait_async0();
  }
  __syncthreads();

  for (int k0 = 0; k0 < C_; k0 += 32) {
    const int  cur = (k0 >> 5) & 1;
    const bool nxt = (k0 + 32) < C_;

    // ---- issue next tile: A async direct to LDS, B via registers ----
    float bv[16];
    if (nxt) {
      const bf16_t* ap = yb + (size_t)(m0 + a_row) * C_ + (k0 + 32) + a_k;
      const unsigned dst = (unsigned)((cur ^ 1) * 8192) + a_lds;
      async_g2l_b128(dst,      ap);
      async_g2l_b128(dst + 16, ap + 8);
      const float* bp = W + (size_t)(k0 + 32 + b_k) * C_ + n0 + b_n;
#pragma unroll
      for (int j = 0; j < 16; ++j) bv[j] = bp[(size_t)j * C_];
    }

    // ---- compute on current buffer ----
    v16bf afr[2];
#pragma unroll
    for (int mi = 0; mi < 2; ++mi) {
      const bf16_t* ar = &Abuf[cur * 4096 + (wm * 32 + mi * 16 + l15) * 32];
      afr[mi] = cat16(*(const v8bf*)(ar + half * 8),
                      *(const v8bf*)(ar + 16 + half * 8));
    }
#pragma unroll
    for (int ni = 0; ni < 4; ++ni) {
      const bf16_t* br = &Bbuf[cur * 4096 + (wn * 64 + ni * 16 + l15) * 32 + half * 8];
      v16bf bfr = cat16(*(const v8bf*)br, *(const v8bf*)(br + 16));
#pragma unroll
      for (int mi = 0; mi < 2; ++mi)
        acc[mi][ni] = wmma_bf16(afr[mi], bfr, acc[mi][ni]);
    }

    // ---- finish staging next tile ----
    if (nxt) {
      v8bf blo, bhi;
#pragma unroll
      for (int j = 0; j < 8; ++j) { blo[j] = (bf16_t)bv[j]; bhi[j] = (bf16_t)bv[8 + j]; }
      *(v8bf*)&Bbuf[(cur ^ 1) * 4096 + b_n * 32 + b_k]     = blo;
      *(v8bf*)&Bbuf[(cur ^ 1) * 4096 + b_n * 32 + b_k + 8] = bhi;
      wait_async0();            // async A writes visible before the barrier
    }
    __syncthreads();
  }

#pragma unroll
  for (int ni = 0; ni < 4; ++ni) {
    const int n = n0 + wn * 64 + ni * 16 + l15;
    const float bvv = bias[n];
#pragma unroll
    for (int mi = 0; mi < 2; ++mi) {
#pragma unroll
      for (int g = 0; g < 8; ++g) {
        const int m = m0 + wm * 32 + mi * 16 + g + 8 * half;
        out[(size_t)m * C_ + n] = acc[mi][ni][g] + bvv;
      }
    }
  }
}

// ---------------------------------------------------------------------------
extern "C" void kernel_launch(void* const* d_in, const int* in_sizes, int n_in,
                              void* d_out, int out_size, void* d_ws, size_t ws_size,
                              hipStream_t stream)
{
  const float* x     = (const float*)d_in[0];
  const float* Wqkv  = (const float*)d_in[1];
  const float* bqkv  = (const float*)d_in[2];
  const float* Wproj = (const float*)d_in[3];
  const float* bproj = (const float*)d_in[4];
  // d_in[5] = prefix_len (128, baked into attn_kernel's mask logic)

  const size_t per = (size_t)B_ * H_ * T_ * D_;   // 4M bf16 elements
  bf16_t* qb  = (bf16_t*)d_ws;
  bf16_t* kb  = qb  + per;
  bf16_t* vtb = kb  + per;
  bf16_t* ybf = vtb + per;                        // [B*T, C] bf16

  // K1: qkv GEMM  (M=4096 x N=3072 tiles of 128x128 -> 32*24 blocks)
  qkv_gemm_kernel<<<dim3(32 * 24), dim3(256), 0, stream>>>(x, Wqkv, bqkv, qb, kb, vtb);

  // K2: attention, one block per (b,h,16-row q block) = 2*16*128 = 4096 blocks
  const size_t smem = (size_t)16 * T_ * 4 + (size_t)16 * T_ * 2 + 16 * 64 * 4 + 64;
  attn_kernel<<<dim3(B_ * H_ * (T_ / 16)), dim3(256), smem, stream>>>(qb, kb, vtb, ybf);

  // K3: output projection (M=4096 x N=1024 -> 32*8 blocks), async-LDS A staging
  proj_gemm_kernel<<<dim3(32 * 8), dim3(256), 32768, stream>>>(ybf, Wproj, bproj,
                                                               (float*)d_out);
}
